// Tnn_Bin_Conv2d_67937792688514
// MI455X (gfx1250) — 8-bit matrix operand order repaired and checked by reading, compiled, not run on hardware
//
#include <hip/hip_runtime.h>
#include <hip/hip_bf16.h>
#include <stdint.h>

#define NB   32
#define C    256
#define H    56
#define W    56
#define HW   (H*W)
#define HP   58
#define KTAP 9
#define KTOT (KTAP*C)
#define MTOT ((long)NB*HW)
#define MSUPER (int)(MTOT/64)
#define NSUPER (C/64)

#define XS_BYTES   ((size_t)NB*HP*HP*C)
#define WSG_BYTES  ((size_t)C*KTAP*C)
#define OFF_XSIGN  ((size_t)0)
#define OFF_WSIGN  (OFF_XSIGN + XS_BYTES)
#define OFF_ALPHA  (OFF_WSIGN + WSG_BYTES)
#define OFF_MEANV  (OFF_ALPHA + 1024)
#define WS_NEEDED  (OFF_MEANV + 2048)

typedef __attribute__((ext_vector_type(8))) int v8i;

__global__ void binact_kernel(const float* __restrict__ x, signed char* __restrict__ xs) {
    long gid = (long)blockIdx.x * blockDim.x + threadIdx.x;
    int  cg  = (int)(gid & 15);
    long pix = gid >> 4;
    int  wp  = (int)(pix % HP);
    int  hp  = (int)((pix / HP) % HP);
    int  n   = (int)(pix / (HP*HP));
    signed char o16[16];
    if (hp == 0 || hp == HP-1 || wp == 0 || wp == HP-1) {
        #pragma unroll
        for (int j = 0; j < 16; ++j) o16[j] = 0;
    } else {
        int h = hp - 1, w = wp - 1;
        const float* p = x + ((long)n*C + (long)cg*16)*HW + (long)h*W + w;
        #pragma unroll
        for (int j = 0; j < 16; ++j) {
            float v = p[(long)j*HW];
            o16[j] = (signed char)((v > 0.0f) - (v < 0.0f));
        }
    }
    *(int4*)(xs + pix*(long)C + (long)cg*16) = *(const int4*)o16;
}

__global__ void wprep_kernel(const float* __restrict__ w,
                             signed char* __restrict__ wsign,
                             float* __restrict__ alphaA) {
    int o  = blockIdx.x;
    int ci = threadIdx.x;
    __shared__ float red[256];
    float v[KTAP];
    const float* p = w + ((long)o*C + ci)*KTAP;
    #pragma unroll
    for (int t = 0; t < KTAP; ++t) v[t] = p[t];

    float aacc = 0.0f;
    signed char sg[KTAP];
    #pragma unroll
    for (int t = 0; t < KTAP; ++t) {
        red[ci] = v[t];
        __syncthreads();
        for (int s = 128; s > 0; s >>= 1) {
            if (ci < s) red[ci] += red[ci + s];
            __syncthreads();
        }
        float mean = red[0] * (1.0f/256.0f);
        __syncthreads();
        float wc = v[t] - mean;
        aacc += fminf(fabsf(wc), 1.0f);
        sg[t] = (signed char)((wc > 0.0f) - (wc < 0.0f));
    }
    red[ci] = aacc;
    __syncthreads();
    for (int s = 128; s > 0; s >>= 1) {
        if (ci < s) red[ci] += red[ci + s];
        __syncthreads();
    }
    if (ci == 0) alphaA[o] = red[0] * (1.0f/2304.0f);
    #pragma unroll
    for (int t = 0; t < KTAP; ++t)
        wsign[((long)o*KTAP + t)*C + ci] = sg[t];
}

__global__ void __launch_bounds__(256)
bconv_wmma_kernel(const signed char* __restrict__ xs,
                  const signed char* __restrict__ wsign,
                  const float* __restrict__ alphaA,
                  const float* __restrict__ biasA,
                  float* __restrict__ out) {
    const int lane = threadIdx.x & 31;
    const int wv   = threadIdx.x >> 5;
    const long tileId = (long)blockIdx.x * 8 + wv;
    const int nsuper = (int)(tileId & (NSUPER - 1));
    const int msuper = (int)(tileId >> 2);
    const int half   = lane >> 4;
    const int l16    = lane & 15;

    const int n_img   = (int)(((long)msuper * 64) / HW);
    const int hw_base = (int)(((long)msuper * 64) % HW);

    const signed char* abase[4];
    #pragma unroll
    for (int s = 0; s < 4; ++s) {
        int hw = hw_base + s*16 + l16;
        int h  = hw / W;
        int ww = hw % W;
        abase[s] = xs + (((long)n_img*HP + h)*HP + ww)*C + half*8;
    }
    const signed char* bbase[4];
    #pragma unroll
    for (int t = 0; t < 4; ++t) {
        int o = nsuper*64 + t*16 + l16;
        bbase[t] = wsign + (long)o*KTAP*C + half*8;
    }

    v8i acc[4][4];
    #pragma unroll
    for (int s = 0; s < 4; ++s)
        #pragma unroll
        for (int t = 0; t < 4; ++t)
            acc[s][t] = (v8i){0,0,0,0,0,0,0,0};

    for (int tap = 0; tap < KTAP; ++tap) {
        const int kh = tap / 3;
        const int kw = tap % 3;
        const long aoff = ((long)kh*HP + kw)*C;
        const long boff = (long)tap*C;

        #pragma unroll
        for (int q = 0; q < 4; ++q) {
            const int cib = q << 6;

            v8i a[4];
            #pragma unroll
            for (int s = 0; s < 4; ++s) {
                const signed char* ap = abase[s] + aoff + cib;
                int2 t0 = *(const int2*)(ap +  0);
                int2 t1 = *(const int2*)(ap + 16);
                int2 t2 = *(const int2*)(ap + 32);
                int2 t3 = *(const int2*)(ap + 48);
                a[s][0]=t0.x; a[s][1]=t0.y; a[s][2]=t1.x; a[s][3]=t1.y;
                a[s][4]=t2.x; a[s][5]=t2.y; a[s][6]=t3.x; a[s][7]=t3.y;
            }
            v8i b[4];
            #pragma unroll
            for (int t = 0; t < 4; ++t) {
                const signed char* bp = bbase[t] + boff + cib;
                __builtin_prefetch(bp + C, 0, 1);
                int2 u0 = *(const int2*)(bp +  0);
                int2 u1 = *(const int2*)(bp + 16);
                int2 u2 = *(const int2*)(bp + 32);
                int2 u3 = *(const int2*)(bp + 48);
                b[t][0]=u0.x; b[t][1]=u0.y; b[t][2]=u1.x; b[t][3]=u1.y;
                b[t][4]=u2.x; b[t][5]=u2.y; b[t][6]=u3.x; b[t][7]=u3.y;
            }

            #pragma unroll
            for (int s = 0; s < 4; ++s)
                #pragma unroll
                for (int t = 0; t < 4; ++t)
                    acc[s][t] = __builtin_amdgcn_wmma_i32_16x16x64_iu8(
                                     true, a[s],  true, b[t], acc[s][t],
                                     false,  false);
        }
    }

    #pragma unroll
    for (int t = 0; t < 4; ++t) {
        const int o = nsuper*64 + t*16 + l16;
        const float alpha = alphaA[o];
        const float bias  = biasA[o];
        float* orow = out + ((long)n_img*C + o)*HW + hw_base;
        #pragma unroll
        for (int s = 0; s < 4; ++s) {
            #pragma unroll
            for (int r = 0; r < 8; ++r) {
                const int mloc = s*16 + r + half*8;
                orow[mloc] = (float)acc[s][t][r] * alpha + bias;
            }
        }
    }
}

__global__ void bn_reduce_kernel(const float* __restrict__ y, float* __restrict__ meanvar) {
    int c = blockIdx.x;
    int t = threadIdx.x;
    float s = 0.0f, s2 = 0.0f;
    for (int n = 0; n < NB; ++n) {
        const float* p = y + ((long)n*C + c)*HW;
        for (int i = t; i < HW; i += 256) {
            float v = p[i];
            s += v; s2 += v*v;
        }
    }
    __shared__ float rs[256], rq[256];
    rs[t] = s; rq[t] = s2;
    __syncthreads();
    for (int k = 128; k > 0; k >>= 1) {
        if (t < k) { rs[t] += rs[t+k]; rq[t] += rq[t+k]; }
        __syncthreads();
    }
    if (t == 0) {
        const float inv = 1.0f / (float)(NB*HW);
        float mean = rs[0] * inv;
        float var  = rq[0] * inv - mean*mean;
        meanvar[c]     = mean;
        meanvar[C + c] = var;
    }
}

__global__ void bn_apply_kernel(float* __restrict__ y,
                                const float* __restrict__ meanvar,
                                const float* __restrict__ gamma,
                                const float* __restrict__ beta) {
    long idx = (long)blockIdx.x * blockDim.x + threadIdx.x;
    long e   = idx * 4;
    int  c   = (int)((e / HW) & (C - 1));
    float mean = meanvar[c];
    float var  = meanvar[C + c];
    float sc   = rsqrtf(var + 1e-5f) * gamma[c];
    float bt   = beta[c];
    float4 v = ((float4*)y)[idx];
    v.x = fmaxf((v.x - mean)*sc + bt, 0.0f);
    v.y = fmaxf((v.y - mean)*sc + bt, 0.0f);
    v.z = fmaxf((v.z - mean)*sc + bt, 0.0f);
    v.w = fmaxf((v.w - mean)*sc + bt, 0.0f);
    ((float4*)y)[idx] = v;
}

extern "C" void kernel_launch(void* const* d_in, const int* in_sizes, int n_in,
                              void* d_out, int out_size, void* d_ws, size_t ws_size,
                              hipStream_t stream) {
    if (ws_size < WS_NEEDED) return;

    const float* x     = (const float*)d_in[0];
    const float* w     = (const float*)d_in[1];
    const float* bias  = (const float*)d_in[2];
    const float* gamma = (const float*)d_in[3];
    const float* beta  = (const float*)d_in[4];
    float* out = (float*)d_out;

    char* ws = (char*)d_ws;
    signed char* xsign = (signed char*)(ws + OFF_XSIGN);
    signed char* wsign = (signed char*)(ws + OFF_WSIGN);
    float* alphaA      = (float*)(ws + OFF_ALPHA);
    float* meanvar     = (float*)(ws + OFF_MEANV);

    {
        long items = (long)NB*HP*HP*16;
        binact_kernel<<<(int)(items/256), 256, 0, stream>>>(x, xsign);
    }
    wprep_kernel<<<C, 256, 0, stream>>>(w, wsign, alphaA);
    bconv_wmma_kernel<<<(MSUPER*NSUPER)/8, 256, 0, stream>>>(
        xsign, wsign, alphaA, bias, out);
    bn_reduce_kernel<<<C, 256, 0, stream>>>(out, meanvar);
    {
        long q = (long)out_size / 4;
        bn_apply_kernel<<<(int)(q/256), 256, 0, stream>>>(out, meanvar, gamma, beta);
    }
}
